// ResidualVQ_81398220193961
// MI455X (gfx1250) — matrix operand order repaired and checked by reading, not compiled, not run on hardware
//
#include <hip/hip_runtime.h>

// ---------------------------------------------------------------------------
// Residual VQ for MI455X (gfx1250, wave32).
// Distance GEMM: v_wmma_f32_16x16x32_bf16, A tile resident in VGPRs, B staged
// block-wide through LDS with double-buffered global_load_async_to_lds_b128
// (ASYNCcnt) so all 8 waves share one copy of each 8 KB codebook chunk.
// Fused argmin keeps the 32768x1024 distance matrix out of memory entirely.
// ---------------------------------------------------------------------------

typedef __bf16 v16bf __attribute__((ext_vector_type(16)));
typedef __bf16 v8bf  __attribute__((ext_vector_type(8)));
typedef float  v8f   __attribute__((ext_vector_type(8)));

static constexpr int Bsz = 64, Tsz = 512, Dd = 256;
static constexpr int Nrows = Bsz * Tsz;          // 32768
static constexpr int Kcb = 1024;                 // codebook size
static constexpr int Qcb = 8;                    // number of codebooks
static constexpr int OUT_ELEMS = Nrows * Dd;     // 8388608
static constexpr int LOSS_OFF  = OUT_ELEMS;      // scalar loss slot
static constexpr int IDX_OFF   = OUT_ELEMS + 1;  // [Q, N] indices (as float)

// ---------------------------------------------------------------------------
// Init: residual = x (f32 + bf16 shadow), zero the loss accumulator.
__global__ void vq_init(const float* __restrict__ x,
                        float* __restrict__ res,
                        __bf16* __restrict__ resb,
                        float* __restrict__ loss) {
  size_t i = ((size_t)blockIdx.x * 256 + threadIdx.x) * 4;
  float4 v = *(const float4*)(x + i);
  *(float4*)(res + i) = v;
  resb[i + 0] = (__bf16)v.x;
  resb[i + 1] = (__bf16)v.y;
  resb[i + 2] = (__bf16)v.z;
  resb[i + 3] = (__bf16)v.w;
  if (blockIdx.x == 0 && threadIdx.x == 0) *loss = 0.0f;
}

// ---------------------------------------------------------------------------
// Per-layer prep: cast codebook to bf16 and compute ||e_k||^2.
__global__ void vq_norms(const float* __restrict__ emb,
                         __bf16* __restrict__ embb,
                         float* __restrict__ colnorm) {
  int k = blockIdx.x, t = threadIdx.x;
  float v = emb[(size_t)k * Dd + t];
  embb[(size_t)k * Dd + t] = (__bf16)v;
  float s = v * v;
#pragma unroll
  for (int m = 16; m >= 1; m >>= 1) s += __shfl_xor(s, m, 32);
  __shared__ float red[8];
  if ((t & 31) == 0) red[t >> 5] = s;
  __syncthreads();
  if (t == 0) {
    float tot = 0.0f;
#pragma unroll
    for (int i = 0; i < 8; ++i) tot += red[i];
    colnorm[k] = tot;
  }
}

// ---------------------------------------------------------------------------
// Fused distance-GEMM + argmin. 8 waves/block, one 16-row tile per wave.
// score(n) = ||e_n||^2 - 2 * <res_row, e_n>   (row norm is argmin-invariant)
__global__ void __launch_bounds__(256)
vq_gemm_argmin(const __bf16* __restrict__ A,      // [N, 256] residual bf16
               const __bf16* __restrict__ Bm,     // [K, 256] codebook bf16
               const float* __restrict__ colnorm, // [K]
               int* __restrict__ idxi,            // [N]
               float* __restrict__ idxf) {        // [N] (d_out slice)
  // Double-buffered B chunk: 16 codewords x 256 K of bf16 = 8 KB per buffer.
  __shared__ __align__(64) __bf16 sB[2][16 * Dd];

  const int lane = threadIdx.x & 31;
  const int wave = threadIdx.x >> 5;
  const int tile = blockIdx.x * 8 + wave;          // 16-row tile id
  const int l15  = lane & 15;
  const bool hi  = lane >= 16;

  // ---- preload A tile, contiguous K order (k = 16h + s) matching B ----
  const __bf16* aRow = A + (size_t)(tile * 16 + l15) * Dd;
  const int asel = hi ? 16 : 0;
  v16bf a[8];
#pragma unroll
  for (int kk = 0; kk < 8; ++kk) {
    v8bf lo = *(const v8bf*)(aRow + kk * 32 + asel);
    v8bf hc = *(const v8bf*)(aRow + kk * 32 + asel + 8);
    v16bf t;
#pragma unroll
    for (int i = 0; i < 8; ++i) { t[i] = lo[i]; t[i + 8] = hc[i]; }
    a[kk] = t;
  }

  // Async stage of one 8 KB contiguous B chunk into LDS buffer `buf`.
  // 256 threads x 32 B each (two b128 transfers per lane), ASYNCcnt-tracked.
  // The LDS destination address is derived from &sB (ptrtoint => the buffer
  // escapes into the asm, so the "memory" clobber makes its writes visible
  // to the compiler and the ds_loads below cannot be folded away).
  auto stage = [&](int c, int buf) {
    unsigned long long src =
        (unsigned long long)(const void*)(Bm + (size_t)c * 4096) +
        (unsigned long long)(threadIdx.x * 16);
    unsigned int dst =
        (unsigned int)(unsigned long long)(const void*)(&sB[buf][0]) +
        (unsigned int)(threadIdx.x * 16);
    asm volatile(
        "global_load_async_to_lds_b128 %0, %1, off\n\t"
        "global_load_async_to_lds_b128 %0, %1, off offset:4096"
        :: "v"(dst), "v"(src) : "memory");
  };

  stage(0, 0);
  asm volatile("s_wait_asynccnt 0x0" ::: "memory");
  __syncthreads();

  float best[8];
  int   bidx[8];
#pragma unroll
  for (int j = 0; j < 8; ++j) { best[j] = 3.4e38f; bidx[j] = 0; }

  for (int c = 0; c < Kcb / 16; ++c) {             // 64 column chunks
    const int buf = c & 1;
    if (c + 1 < Kcb / 16) stage(c + 1, buf ^ 1);   // prefetch next chunk

    // B lane view: lane n<16 holds col n, K 0..15; lane>=16 col n-16, K 16..31
    const __bf16* bp = &sB[buf][l15 * Dd + (hi ? 16 : 0)];
    // Two independent accumulator chains so successive WMMAs can pipeline.
    v8f acc0 = {};
    v8f acc1 = {};
#pragma unroll
    for (int kk = 0; kk < 4; ++kk) {
      v16bf b0 = *(const v16bf*)(bp + kk * 32);            // ds_load_b128 x2
      v16bf b1 = *(const v16bf*)(bp + (kk + 4) * 32);
      acc0 = __builtin_amdgcn_wmma_f32_16x16x32_bf16(
          false, a[kk], false, b0, (short)0, acc0, false, false);
      acc1 = __builtin_amdgcn_wmma_f32_16x16x32_bf16(
          false, a[kk + 4], false, b1, (short)0, acc1, false, false);
    }
    const int n = c * 16 + l15;                    // this lane's column
    const float cn = colnorm[n];
#pragma unroll
    for (int j = 0; j < 8; ++j) {                  // C: vgpr j = row (j or 8+j)
      float s = cn - 2.0f * (acc0[j] + acc1[j]);
      if (s < best[j]) { best[j] = s; bidx[j] = n; }
    }

    asm volatile("s_wait_asynccnt 0x0" ::: "memory");
    __syncthreads();
  }

  // ---- cross-lane min+index reduction over the 16 lanes sharing a row ----
#pragma unroll
  for (int j = 0; j < 8; ++j) {
    float s = best[j];
    int   bi = bidx[j];
#pragma unroll
    for (int m = 1; m < 16; m <<= 1) {
      float os = __shfl_xor(s, m, 32);
      int   oi = __shfl_xor(bi, m, 32);
      if (os < s || (os == s && oi < bi)) { s = os; bi = oi; }
    }
    if (l15 == 0) {
      int row = tile * 16 + (hi ? (8 + j) : j);
      idxi[row] = bi;
      idxf[row] = (float)bi;
    }
  }
}

// ---------------------------------------------------------------------------
// Gather q = emb[idx], loss += 1.25*mean((q-res)^2), res -= q (f32 + bf16).
__global__ void vq_update(const float* __restrict__ emb,
                          const int* __restrict__ idx,
                          float* __restrict__ res,
                          __bf16* __restrict__ resb,
                          float* __restrict__ loss) {
  const int row = blockIdx.x * 4 + (threadIdx.x >> 6);
  const int d   = (threadIdx.x & 63) * 4;
  const int k   = idx[row];
  const float4 q = *(const float4*)(emb + (size_t)k * Dd + d);
  float* rp = res + (size_t)row * Dd + d;
  float4 r = *(const float4*)rp;
  float dx = q.x - r.x, dy = q.y - r.y, dz = q.z - r.z, dw = q.w - r.w;
  float s = dx * dx + dy * dy + dz * dz + dw * dw;
  float4 rn = make_float4(r.x - q.x, r.y - q.y, r.z - q.z, r.w - q.w);
  *(float4*)rp = rn;
  __bf16* bp = resb + (size_t)row * Dd + d;
  bp[0] = (__bf16)rn.x; bp[1] = (__bf16)rn.y;
  bp[2] = (__bf16)rn.z; bp[3] = (__bf16)rn.w;
#pragma unroll
  for (int m = 16; m >= 1; m >>= 1) s += __shfl_xor(s, m, 32);
  __shared__ float red[8];
  if ((threadIdx.x & 31) == 0) red[threadIdx.x >> 5] = s;
  __syncthreads();
  if (threadIdx.x == 0) {
    float tot = 0.0f;
#pragma unroll
    for (int i = 0; i < 8; ++i) tot += red[i];
    atomicAdd(loss, tot * (1.25f / ((float)Nrows * (float)Dd)));
  }
}

// ---------------------------------------------------------------------------
// output = x - final residual
__global__ void vq_finalize(const float* __restrict__ x,
                            const float* __restrict__ res,
                            float* __restrict__ out) {
  size_t i = ((size_t)blockIdx.x * 256 + threadIdx.x) * 4;
  float4 xv = *(const float4*)(x + i);
  float4 rv = *(const float4*)(res + i);
  *(float4*)(out + i) = make_float4(xv.x - rv.x, xv.y - rv.y,
                                    xv.z - rv.z, xv.w - rv.w);
}

// ---------------------------------------------------------------------------
extern "C" void kernel_launch(void* const* d_in, const int* in_sizes, int n_in,
                              void* d_out, int out_size, void* d_ws, size_t ws_size,
                              hipStream_t stream) {
  (void)in_sizes; (void)n_in; (void)out_size; (void)ws_size;
  const float* x   = (const float*)d_in[0];
  const float* emb = (const float*)d_in[1];   // [Q, K, D]
  float* out = (float*)d_out;

  // workspace carve (~48.6 MB)
  char* w = (char*)d_ws;
  float*  res_f32  = (float*)(w);                                 // 33554432 B
  __bf16* res_bf16 = (__bf16*)(w + 33554432);                     // 16777216 B
  __bf16* emb_bf16 = (__bf16*)(w + 50331648);                     //   524288 B
  float*  colnorm  = (float*)(w + 50855936);                      //     4096 B
  int*    idx_ws   = (int*)(w + 50860032);                        //   131072 B

  float* loss_ptr = out + LOSS_OFF;

  vq_init<<<8192, 256, 0, stream>>>(x, res_f32, res_bf16, loss_ptr);

  for (int l = 0; l < Qcb; ++l) {
    const float* emb_l = emb + (size_t)l * Kcb * Dd;
    vq_norms<<<Kcb, Dd, 0, stream>>>(emb_l, emb_bf16, colnorm);
    vq_gemm_argmin<<<Nrows / 16 / 8, 256, 0, stream>>>(
        res_bf16, emb_bf16, colnorm, idx_ws, out + IDX_OFF + (size_t)l * Nrows);
    vq_update<<<Nrows / 4, 256, 0, stream>>>(emb_l, idx_ws, res_f32, res_bf16,
                                             loss_ptr);
  }

  vq_finalize<<<8192, 256, 0, stream>>>(x, res_f32, out);
}
